// RetainEx_13185549598731
// MI455X (gfx1250) — matrix operand order repaired and checked by reading, not compiled, not run on hardware
//
#include <hip/hip_runtime.h>
#include <hip/hip_bf16.h>
#include <math.h>

// ---------------- types ----------------
typedef __bf16 bf16_t;
typedef bf16_t v16bf __attribute__((ext_vector_type(16)));
typedef bf16_t v8bf  __attribute__((ext_vector_type(8)));
typedef float  v8f   __attribute__((ext_vector_type(8)));

#define Bz   128
#define Sz   64
#define Cz   20
#define Dz   256
#define KPAD 288          // D+3 padded up to multiple of 32
#define G4   1024         // 4*D
#define TD   128          // TDIM
#define NST  1000

__device__ __forceinline__ bf16_t f2bf(float f) {
    union { float f; unsigned u; } v; v.f = f;
    unsigned r = (v.u + 0x7FFFu + ((v.u >> 16) & 1u)) >> 16;
    union { unsigned short s; bf16_t b; } o; o.s = (unsigned short)r;
    return o.b;
}
__device__ __forceinline__ float sigmf(float x) { return 1.0f / (1.0f + expf(-x)); }

// ---------------- WMMA GEMM: C[M,N] = A[M,K](bf16,row) * Bt[N,K](bf16,row)^T (+Cin)(+bias)
// Requires: M%16==0, N%16==0, K%32==0. One wave computes one 16x16 tile.
__global__ void wmma_gemm_bf16(
    const bf16_t* __restrict__ A,  long sAz,
    const bf16_t* __restrict__ Bt, long sBz,
    float* __restrict__ C,         long sCz,
    const float* __restrict__ Cin, long sCinz, int ldcin,
    const float* __restrict__ bias,long sBiasz,
    int M, int N, int K, int lda, int ldb, int ldc)
{
    int z = blockIdx.z;
    A  += z * sAz;  Bt += z * sBz;  C += z * sCz;
    if (Cin)  Cin  += z * sCinz;
    if (bias) bias += z * sBiasz;

    int tiles_n = N >> 4;
    int tiles   = (M >> 4) * tiles_n;
    int wave = (blockIdx.x * blockDim.x + threadIdx.x) >> 5;
    if (wave >= tiles) return;
    int tm = wave / tiles_n, tn = wave % tiles_n;

    int lane = threadIdx.x & 31;
    int half = lane >> 4;          // 0: K low halves, 1: K high halves
    int r    = lane & 15;

    const bf16_t* arow = A  + (long)(tm * 16 + r) * lda + half * 16;
    const bf16_t* brow = Bt + (long)(tn * 16 + r) * ldb + half * 16;

    v8f acc = {};
    for (int k0 = 0; k0 < K; k0 += 32) {
        __builtin_prefetch(arow + k0 + 128, 0, 1);
        __builtin_prefetch(brow + k0 + 128, 0, 1);
        v16bf afrag = *(const v16bf*)(arow + k0);    // K = k0 + half*16 + 0..15
        v16bf bfrag = *(const v16bf*)(brow + k0);    // K = k0 + half*16 + 0..15
        acc = __builtin_amdgcn_wmma_f32_16x16x32_bf16(
            false, afrag, false, bfrag, (short)0, acc, false, false);
    }

    int col = tn * 16 + r;
    float bv = bias ? bias[col] : 0.0f;
#pragma unroll
    for (int i = 0; i < 8; i++) {
        int row = tm * 16 + i + half * 8;
        float v = acc[i] + bv;
        if (Cin) v += Cin[(long)row * ldcin + col];
        C[(long)row * ldc + col] = v;
    }
}

// ---------------- fused LSTM recurrent step ----------------
// One wave computes a 16x16 (batch x hidden-j) tile for ALL FOUR gate blocks:
// 4 accumulators share one A (=h) fragment, 4 WMMAs per K-chunk, then the
// LSTM cell update happens per-lane in the epilogue (i/f/g/o live in the same
// lane/element position of acc0..acc3). h ping-pongs across steps.
__global__ void lstm_step_fused(
    const bf16_t* __restrict__ hin,   // (2,B,D) bf16   (read)
    const bf16_t* __restrict__ Whhp,  // (2,G4,D) bf16
    const float*  __restrict__ gatesI,// (2,B*S,G4) input-projection gates
    const float*  __restrict__ bias2, // (2,G4)
    float* __restrict__ cst,          // (2,B,D) cell state (in-place)
    bf16_t* __restrict__ hout,        // (2,B,D) bf16   (write)
    float* __restrict__ out0,         // hidden-seq out, z-stride outz
    long outz, int t)
{
    int wave = (blockIdx.x * blockDim.x + threadIdx.x) >> 5;   // 0..255
    if (wave >= 2 * (Bz / 16) * (Dz / 16)) return;
    int z  = wave >> 7;            // rnn index
    int w  = wave & 127;
    int tm = w >> 4;               // batch tile 0..7
    int tn = w & 15;               // j tile 0..15

    int lane = threadIdx.x & 31;
    int half = lane >> 4, r = lane & 15;

    const bf16_t* arow = hin  + (long)z * Bz * Dz + (long)(tm * 16 + r) * Dz + half * 16;
    const bf16_t* b0   = Whhp + (long)z * G4 * Dz + (long)(tn * 16 + r) * Dz + half * 16;

    v8f acc0 = {}, acc1 = {}, acc2 = {}, acc3 = {};
    for (int k0 = 0; k0 < Dz; k0 += 32) {
        v16bf afrag = *(const v16bf*)(arow + k0);
        v16bf bf0 = *(const v16bf*)(b0 + k0);
        v16bf bf1 = *(const v16bf*)(b0 + (long)256 * Dz + k0);
        v16bf bf2 = *(const v16bf*)(b0 + (long)512 * Dz + k0);
        v16bf bf3 = *(const v16bf*)(b0 + (long)768 * Dz + k0);
        acc0 = __builtin_amdgcn_wmma_f32_16x16x32_bf16(false, afrag, false, bf0, (short)0, acc0, false, false);
        acc1 = __builtin_amdgcn_wmma_f32_16x16x32_bf16(false, afrag, false, bf1, (short)0, acc1, false, false);
        acc2 = __builtin_amdgcn_wmma_f32_16x16x32_bf16(false, afrag, false, bf2, (short)0, acc2, false, false);
        acc3 = __builtin_amdgcn_wmma_f32_16x16x32_bf16(false, afrag, false, bf3, (short)0, acc3, false, false);
    }

    int col = tn * 16 + r;
    const float* cin = gatesI + (long)z * ((long)Bz * Sz) * G4 + (long)t * G4;
    const float* bz  = bias2 + (long)z * G4;
    float*  cbase = cst  + (long)z * Bz * Dz;
    bf16_t* hb    = hout + (long)z * Bz * Dz;
    float*  ob    = out0 + (long)z * outz;

    float b_i = bz[col], b_f = bz[256 + col], b_g = bz[512 + col], b_o = bz[768 + col];
#pragma unroll
    for (int i = 0; i < 8; i++) {
        int row = tm * 16 + i + half * 8;
        long cr = (long)row * (Sz * G4) + t * 0;     // row base into gatesI (t folded in cin)
        const float* g = cin + (long)row * (Sz * G4);
        float gi = acc0[i] + b_i + g[col];
        float gf = acc1[i] + b_f + g[256 + col];
        float gg = acc2[i] + b_g + g[512 + col];
        float go = acc3[i] + b_o + g[768 + col];
        (void)cr;
        long ci = (long)row * Dz + col;
        float cn = sigmf(gf) * cbase[ci] + sigmf(gi) * tanhf(gg);
        float h  = sigmf(go) * tanhf(cn);
        cbase[ci] = cn;
        hb[ci]    = f2bf(h);
        ob[((long)row * Sz + t) * Dz + col] = h;
    }
}

// ---------------- packing / conversion ----------------
__global__ void pack_weight(const float* __restrict__ src, bf16_t* __restrict__ dst,
                            int Nr, int Kc, int ldb, int transpose)
{
    long i = (long)blockIdx.x * blockDim.x + threadIdx.x;
    long total = (long)Nr * ldb;
    if (i >= total) return;
    int n = (int)(i / ldb), k = (int)(i % ldb);
    float v = 0.0f;
    if (k < Kc) v = transpose ? src[(long)k * Nr + n] : src[(long)n * Kc + k];
    dst[i] = f2bf(v);
}

__global__ void cvt_bf16(const float* __restrict__ src, bf16_t* __restrict__ dst, long n)
{
    long i = (long)blockIdx.x * blockDim.x + threadIdx.x;
    if (i < n) dst[i] = f2bf(src[i]);
}

__global__ void concat_bias(const float* __restrict__ a, const float* __restrict__ b,
                            float* __restrict__ dst, int n)
{
    int i = blockIdx.x * blockDim.x + threadIdx.x;
    if (i < 2 * n) dst[i] = (i < n) ? a[i] : b[i - n];
}

// ---------------- embedding gather + dates + bf16 pack ----------------
__global__ void embed_kernel(const int* __restrict__ seqs, const float* __restrict__ tstep,
                             const float* __restrict__ emb1, const float* __restrict__ emb2,
                             float* __restrict__ embf, float* __restrict__ emb2vf,
                             float* __restrict__ dates, bf16_t* __restrict__ Xp)
{
    long i = (long)blockIdx.x * blockDim.x + threadIdx.x;
    if (i >= (long)Bz * Sz * KPAD) return;
    int bs = (int)(i / KPAD), d = (int)(i % KPAD);
    if (d < Dz) {
        const int* row = seqs + (long)bs * Cz;
        float e1 = 0.0f, e2 = 0.0f;
        for (int c = 0; c < Cz; c++) {
            int id = row[c];
            e1 += emb1[(long)id * Dz + d];
            e2 += emb2[(long)id * Dz + d];
        }
        embf[(long)bs * Dz + d]   = e1;
        emb2vf[(long)bs * Dz + d] = e2;
        Xp[i] = f2bf(e1);
    } else if (d < Dz + 3) {
        float t = tstep[bs];
        float v = (d == Dz) ? t : (d == Dz + 1) ? 1.0f / (t + 1.0f)
                                                : 1.0f / logf(2.718281828459045f + t);
        dates[(long)bs * 3 + (d - Dz)] = v;
        Xp[i] = f2bf(v);
    } else {
        Xp[i] = f2bf(0.0f);
    }
}

// ---------------- LSTM init ----------------
__global__ void lstm_init(bf16_t* __restrict__ hbf, float* __restrict__ c)
{
    int i = blockIdx.x * blockDim.x + threadIdx.x;
    if (i < 2 * Bz * Dz) { hbf[i] = f2bf(0.0f); c[i] = 0.0f; }
}

// ---------------- alignment attention ----------------
__global__ void copy_row0(const float* __restrict__ embf, float* __restrict__ aligned)
{
    int i = blockIdx.x * blockDim.x + threadIdx.x;   // B*D
    if (i >= Bz * Dz) return;
    int b = i >> 8, d = i & 255;
    aligned[((long)b * Sz) * Dz + d] = embf[((long)b * Sz) * Dz + d];
}

__global__ void align_kernel(const float* __restrict__ embf, const float* __restrict__ whk_full,
                             const float* __restrict__ w1w, const float* __restrict__ w1b,
                             const float* __restrict__ w2w, const float* __restrict__ w2b,
                             float* __restrict__ aligned)
{
    int b  = blockIdx.x / (Sz - 1);
    int sp = blockIdx.x % (Sz - 1);                  // 0..62
    __shared__ float ek[Dz], wh[Dz], hb[64], lg[2];
    int t = threadIdx.x;                             // 64 threads
    for (int d = t; d < Dz; d += 64) {
        ek[d] = embf[((long)b * Sz) * Dz + d];
        wh[d] = whk_full[((long)b * Sz + sp) * Dz + d];
    }
    __syncthreads();
    float acc = w1b[t];
    for (int k = 0; k < Dz; k++) acc += ek[k] * w1w[k * 64 + t];
    for (int k = 0; k < Dz; k++) acc += wh[k] * w1w[(Dz + k) * 64 + t];
    hb[t] = tanhf(acc);
    __syncthreads();
    if (t == 0) {
        float l0 = w2b[0], l1 = w2b[1];
        for (int j = 0; j < 64; j++) { l0 += hb[j] * w2w[j * 2]; l1 += hb[j] * w2w[j * 2 + 1]; }
        float mx = fmaxf(l0, l1);
        float e0 = expf(l0 - mx), e1 = expf(l1 - mx), s = e0 + e1;
        lg[0] = e0 / s; lg[1] = e1 / s;
    }
    __syncthreads();
    float a0 = lg[0], a1 = lg[1];
    for (int d = t; d < Dz; d += 64)
        aligned[((long)b * Sz + sp + 1) * Dz + d] = ek[d] * a0 + wh[d] * a1;
}

// ---------------- diffusion ----------------
__global__ void cumprod_kernel(const float* __restrict__ betas, float* __restrict__ ab)
{
    if (blockIdx.x == 0 && threadIdx.x == 0) {
        float p = 1.0f;
        for (int i = 0; i < NST; i++) { p *= (1.0f - betas[i]); ab[i] = p; }
    }
}

__global__ void noisy_kernel(const float* __restrict__ aligned, const float* __restrict__ noise,
                             const int* __restrict__ tt, const float* __restrict__ ab,
                             float* __restrict__ noisy)
{
    long i = (long)blockIdx.x * blockDim.x + threadIdx.x;
    if (i >= (long)Bz * Sz * Dz) return;
    int b = (int)(i / ((long)Sz * Dz));
    float a = ab[tt[b]];
    noisy[i] = aligned[i] * sqrtf(a) + noise[i] * sqrtf(1.0f - a);
}

__global__ void temb_kernel(const int* __restrict__ tt, const float* __restrict__ tw,
                            const float* __restrict__ tb, float* __restrict__ tout)
{
    int b = blockIdx.x, j = threadIdx.x;             // TD threads
    __shared__ float te[TD];
    float t = (float)tt[b];
    float lg = logf(10000.0f);
    if (j < 64) te[j] = cosf(t * expf(-lg * (float)j / 64.0f));
    else        te[j] = sinf(t * expf(-lg * (float)(j - 64) / 64.0f));
    __syncthreads();
    float acc = tb[j];
    for (int k = 0; k < TD; k++) acc += te[k] * tw[k * TD + j];
    tout[(long)b * TD + j] = acc * sigmf(acc);       // silu
}

__global__ void im2col1_kernel(const float* __restrict__ noisy, bf16_t* __restrict__ dst)
{
    long i = (long)blockIdx.x * blockDim.x + threadIdx.x;   // 32768*192
    if (i >= (long)Bz * Dz * 192) return;
    int m = (int)(i / 192), col = (int)(i % 192);
    int b = m >> 8, l = m & 255;
    int ic = col / 3, tp = col % 3;
    int pos = l + tp - 1;
    float v = (pos >= 0 && pos < Dz) ? noisy[((long)b * Sz + ic) * Dz + pos] : 0.0f;
    dst[i] = f2bf(v);
}

__global__ void im2col2_kernel(const float* __restrict__ C1, const float* __restrict__ tembo,
                               bf16_t* __restrict__ dst)
{
    long i = (long)blockIdx.x * blockDim.x + threadIdx.x;   // 32768*384
    if (i >= (long)Bz * Dz * 384) return;
    int m = (int)(i / 384), col = (int)(i % 384);
    int b = m >> 8, l = m & 255;
    int ic = col / 3, tp = col % 3;
    int pos = l + tp - 1;
    float v = 0.0f;
    if (pos >= 0 && pos < Dz) {
        float h = C1[((long)b * Dz + pos) * TD + ic] + tembo[(long)b * TD + ic];
        v = h * sigmf(h);                                  // silu (conv1_b added in GEMM)
    }
    dst[i] = f2bf(v);
}

__global__ void gen_kernel(const float* __restrict__ C2, const float* __restrict__ aligned,
                           const float* __restrict__ noise, const float* __restrict__ dates,
                           float* __restrict__ pred_out, bf16_t* __restrict__ gxp)
{
    long i = (long)blockIdx.x * blockDim.x + threadIdx.x;   // 8192*288
    if (i >= (long)Bz * Sz * KPAD) return;
    int bs = (int)(i / KPAD), d = (int)(i % KPAD);
    int b = bs >> 6, s = bs & 63;
    if (d < Dz) {
        float p = C2[((long)b * Dz + d) * Sz + s];          // conv2_b added in GEMM
        long pi = (long)bs * Dz + d;
        pred_out[pi] = p;
        gxp[i] = f2bf(aligned[pi] + noise[pi] - p);
    } else if (d < Dz + 3) {
        gxp[i] = f2bf(dates[(long)bs * 3 + (d - Dz)]);
    } else {
        gxp[i] = f2bf(0.0f);
    }
}

// ---------------- final attend ----------------
__global__ void attend_kernel(const float* __restrict__ o1, const float* __restrict__ beta,
                              const float* __restrict__ emb2vf, const float* __restrict__ wa,
                              const float* __restrict__ ow, const float* __restrict__ ob,
                              float* __restrict__ outv)
{
    int b = blockIdx.x;
    __shared__ float alpha[Sz];
    __shared__ float red[256];
    int t = threadIdx.x;                                    // 256 threads
    if (t < Sz) {
        float acc = 0.0f;
        const float* row = o1 + ((long)b * Sz + t) * Dz;
        for (int k = 0; k < Dz; k++) acc += row[k] * wa[k];
        alpha[t] = acc;
    }
    __syncthreads();
    if (t == 0) {
        float mx = -1e30f;
        for (int s = 0; s < Sz; s++) mx = fmaxf(mx, alpha[s]);
        float sum = 0.0f;
        for (int s = 0; s < Sz; s++) { alpha[s] = expf(alpha[s] - mx); sum += alpha[s]; }
        for (int s = 0; s < Sz; s++) alpha[s] /= sum;
    }
    __syncthreads();
    float v = 0.0f;
    for (int s = 0; s < Sz; s++) {
        long idx = ((long)b * Sz + s) * Dz + t;
        v += emb2vf[idx] * tanhf(beta[idx]) * alpha[s];
    }
    red[t] = v * ow[t * 2 + 0];
    __syncthreads();
    for (int st = 128; st > 0; st >>= 1) { if (t < st) red[t] += red[t + st]; __syncthreads(); }
    if (t == 0) outv[b * 2 + 0] = red[0] + ob[0];
    __syncthreads();
    red[t] = v * ow[t * 2 + 1];
    __syncthreads();
    for (int st = 128; st > 0; st >>= 1) { if (t < st) red[t] += red[t + st]; __syncthreads(); }
    if (t == 0) outv[b * 2 + 1] = red[0] + ob[1];
}

// ---------------- host orchestration ----------------
static void gemm(hipStream_t s,
                 const bf16_t* A, long sAz, const bf16_t* Bt, long sBz,
                 float* C, long sCz, const float* Cin, long sCinz, int ldcin,
                 const float* bias, long sBiasz,
                 int M, int N, int K, int lda, int ldb, int ldc, int nz)
{
    int tiles = (M / 16) * (N / 16);
    dim3 g((tiles + 7) / 8, 1, nz);
    wmma_gemm_bf16<<<g, dim3(256), 0, s>>>(A, sAz, Bt, sBz, C, sCz,
                                           Cin, sCinz, ldcin, bias, sBiasz,
                                           M, N, K, lda, ldb, ldc);
}

#define EL(kern, n, ...) kern<<<dim3((int)(((long)(n) + 255) / 256)), dim3(256), 0, stream>>>(__VA_ARGS__)

extern "C" void kernel_launch(void* const* d_in, const int* in_sizes, int n_in,
                              void* d_out, int out_size, void* d_ws, size_t ws_size,
                              hipStream_t stream)
{
    const int*   seqs   = (const int*)d_in[0];
    const float* tstep  = (const float*)d_in[3];
    const int*   dt     = (const int*)d_in[5];
    const float* noise  = (const float*)d_in[6];
    const float* emb1   = (const float*)d_in[7];
    const float* emb2w  = (const float*)d_in[8];
    const float* r1_wih = (const float*)d_in[9];
    const float* r1_whh = (const float*)d_in[10];
    const float* r1_b   = (const float*)d_in[11];
    const float* r2_wih = (const float*)d_in[12];
    const float* r2_whh = (const float*)d_in[13];
    const float* r2_b   = (const float*)d_in[14];
    const float* wa_w   = (const float*)d_in[15];
    const float* Wb_w   = (const float*)d_in[16];
    const float* whk_w  = (const float*)d_in[17];
    const float* whk_b  = (const float*)d_in[18];
    const float* w1_w   = (const float*)d_in[19];
    const float* w1_b   = (const float*)d_in[20];
    const float* w2_w   = (const float*)d_in[21];
    const float* w2_b   = (const float*)d_in[22];
    const float* out_w  = (const float*)d_in[23];
    const float* out_b  = (const float*)d_in[24];
    const float* temb_w = (const float*)d_in[25];
    const float* temb_b = (const float*)d_in[26];
    const float* c1k    = (const float*)d_in[27];
    const float* c1b    = (const float*)d_in[28];
    const float* c2k    = (const float*)d_in[29];
    const float* c2b    = (const float*)d_in[30];
    const float* betas  = (const float*)d_in[31];

    float* out_f   = (float*)d_out;                 // [0..255] outputs, [256..511] gen_outputs
    float* pred_o  = out_f + 512;                   // B*S*D
    float* noise_o = pred_o + (long)Bz * Sz * Dz;   // B*S*D

    char* ws = (char*)d_ws;
    size_t off = 0;
    auto alloc = [&](size_t bytes) -> char* {
        char* p = ws + off;
        off = (off + bytes + 255) & ~(size_t)255;
        return p;
    };

    const long BS = (long)Bz * Sz;                  // 8192
    bf16_t* Xp     = (bf16_t*)alloc(BS * KPAD * 2);
    bf16_t* Wihp   = (bf16_t*)alloc(2L * G4 * KPAD * 2);
    bf16_t* Whhp   = (bf16_t*)alloc(2L * G4 * Dz * 2);
    float*  bias2  = (float*) alloc(2L * G4 * 4);
    bf16_t* whkp   = (bf16_t*)alloc((long)Dz * Dz * 2);
    bf16_t* Wbp    = (bf16_t*)alloc((long)Dz * Dz * 2);
    bf16_t* c1p    = (bf16_t*)alloc(128L * 192 * 2);
    bf16_t* c2p    = (bf16_t*)alloc(64L * 384 * 2);
    float*  embf   = (float*) alloc(BS * Dz * 4);
    float*  emb2vf = (float*) alloc(BS * Dz * 4);
    float*  datesf = (float*) alloc(BS * 3 * 4);
    float*  gatesI = (float*) alloc(2L * BS * G4 * 4);   // 67 MB; aliased as conv scratch
    bf16_t* hping  = (bf16_t*)alloc(2L * Bz * Dz * 2);   // h double-buffer A
    bf16_t* hpong  = (bf16_t*)alloc(2L * Bz * Dz * 2);   // h double-buffer B
    float*  cst    = (float*) alloc(2L * Bz * Dz * 4);
    float*  out12  = (float*) alloc(2L * BS * Dz * 4);   // out1, out2
    float*  g12    = (float*) alloc(2L * BS * Dz * 4);   // g1, g2
    bf16_t* o1bf   = (bf16_t*)alloc(BS * Dz * 2);
    bf16_t* o2bf   = (bf16_t*)alloc(BS * Dz * 2);
    float*  whkF   = (float*) alloc(BS * Dz * 4);        // also reused as beta buffer
    float*  aligned= (float*) alloc(BS * Dz * 4);
    float*  abbar  = (float*) alloc(NST * 4);
    float*  noisy  = (float*) alloc(BS * Dz * 4);
    float*  tembo  = (float*) alloc((long)Bz * TD * 4);
    bf16_t* gxp    = (bf16_t*)alloc(BS * KPAD * 2);
    (void)ws_size; (void)in_sizes; (void)n_in; (void)out_size;

    // conv scratch aliases gatesI (dead between LSTM pass 1 and pass 2): 63 MB < 67 MB
    char*   cs  = (char*)gatesI;
    bf16_t* im1 = (bf16_t*)cs;                               // 32768*192 bf16 = 12.58 MB
    float*  C1  = (float*)(cs + 12582912);                   // 32768*128 f32  = 16.78 MB
    bf16_t* im2 = (bf16_t*)(cs + 12582912 + 16777216);       // 32768*384 bf16 = 25.17 MB
    float*  C2  = (float*)(cs + 12582912 + 16777216 + 25165824); // 32768*64 f32 = 8.39 MB

    float* out1 = out12;
    float* out2 = out12 + BS * Dz;
    float* g1   = g12;
    float* g2   = g12 + BS * Dz;
    bf16_t* hbufs[2] = { hping, hpong };

    // ---- weight packing (bf16, N x K row-major) ----
    EL(pack_weight, (long)G4 * KPAD, r1_wih, Wihp,                 G4, Dz + 3, KPAD, 0);
    EL(pack_weight, (long)G4 * KPAD, r2_wih, Wihp + (long)G4*KPAD, G4, Dz + 3, KPAD, 0);
    EL(pack_weight, (long)G4 * Dz,   r1_whh, Whhp,                 G4, Dz, Dz, 0);
    EL(pack_weight, (long)G4 * Dz,   r2_whh, Whhp + (long)G4*Dz,   G4, Dz, Dz, 0);
    EL(concat_bias, 2 * G4, r1_b, r2_b, bias2, G4);
    EL(pack_weight, (long)Dz * Dz, whk_w, whkp, Dz, Dz, Dz, 1);    // transpose: (K,N)->(N,K)
    EL(pack_weight, (long)Dz * Dz, Wb_w,  Wbp,  Dz, Dz, Dz, 1);
    EL(pack_weight, 128L * 192, c1k, c1p, 128, 192, 192, 0);
    EL(pack_weight, 64L * 384,  c2k, c2p, 64,  384, 384, 0);

    // ---- embeddings, dates, X (bf16 padded) ----
    EL(embed_kernel, BS * KPAD, seqs, tstep, emb1, emb2w, embf, emb2vf, datesf, Xp);

    // ---- LSTM pass 1: hoisted input GEMM for both RNNs, then fused steps ----
    gemm(stream, Xp, 0, Wihp, (long)G4 * KPAD, gatesI, (long)BS * G4,
         nullptr, 0, 0, nullptr, 0, (int)BS, G4, KPAD, KPAD, KPAD, G4, 2);
    EL(lstm_init, 2 * Bz * Dz, hping, cst);
    for (int t = 0; t < Sz; t++) {
        lstm_step_fused<<<dim3(32), dim3(256), 0, stream>>>(
            hbufs[t & 1], Whhp, gatesI, bias2, cst, hbufs[1 - (t & 1)],
            out12, BS * Dz, t);
    }

    // ---- alignment attention ----
    EL(cvt_bf16, BS * Dz, out1, o1bf, BS * Dz);
    gemm(stream, o1bf, 0, whkp, 0, whkF, 0, nullptr, 0, 0, whk_b, 0,
         (int)BS, Dz, Dz, Dz, Dz, Dz, 1);
    EL(copy_row0, Bz * Dz, embf, aligned);
    align_kernel<<<dim3(Bz * (Sz - 1)), dim3(64), 0, stream>>>(
        embf, whkF, w1_w, w1_b, w2_w, w2_b, aligned);

    // ---- diffusion noising ----
    cumprod_kernel<<<dim3(1), dim3(32), 0, stream>>>(betas, abbar);
    EL(noisy_kernel, BS * Dz, aligned, noise, dt, abbar, noisy);

    // ---- UNet: temb, conv1, conv2 (im2col + WMMA GEMM) ----
    temb_kernel<<<dim3(Bz), dim3(TD), 0, stream>>>(dt, temb_w, temb_b, tembo);
    EL(im2col1_kernel, (long)Bz * Dz * 192, noisy, im1);
    gemm(stream, im1, 0, c1p, 0, C1, 0, nullptr, 0, 0, c1b, 0,
         Bz * Dz, TD, 192, 192, 192, TD, 1);
    EL(im2col2_kernel, (long)Bz * Dz * 384, C1, tembo, im2);
    gemm(stream, im2, 0, c2p, 0, C2, 0, nullptr, 0, 0, c2b, 0,
         Bz * Dz, Sz, 384, 384, 384, Sz, 1);

    // ---- pred out, gen, gx (bf16 padded) ----
    EL(gen_kernel, BS * KPAD, C2, aligned, noise, datesf, pred_o, gxp);
    hipMemcpyAsync(noise_o, noise, (size_t)BS * Dz * sizeof(float),
                   hipMemcpyDeviceToDevice, stream);

    // ---- LSTM pass 2 on gx (overwrites gatesI; conv scratch is dead now) ----
    gemm(stream, gxp, 0, Wihp, (long)G4 * KPAD, gatesI, (long)BS * G4,
         nullptr, 0, 0, nullptr, 0, (int)BS, G4, KPAD, KPAD, KPAD, G4, 2);
    EL(lstm_init, 2 * Bz * Dz, hping, cst);
    for (int t = 0; t < Sz; t++) {
        lstm_step_fused<<<dim3(32), dim3(256), 0, stream>>>(
            hbufs[t & 1], Whhp, gatesI, bias2, cst, hbufs[1 - (t & 1)],
            g12, BS * Dz, t);
    }

    // ---- final attends (beta = o2 @ Wb via WMMA; whkF reused as beta buffer) ----
    EL(cvt_bf16, BS * Dz, out2, o2bf, BS * Dz);
    gemm(stream, o2bf, 0, Wbp, 0, whkF, 0, nullptr, 0, 0, nullptr, 0,
         (int)BS, Dz, Dz, Dz, Dz, Dz, 1);
    attend_kernel<<<dim3(Bz), dim3(256), 0, stream>>>(
        out1, whkF, emb2vf, wa_w, out_w, out_b, out_f);

    EL(cvt_bf16, BS * Dz, g2, o2bf, BS * Dz);
    gemm(stream, o2bf, 0, Wbp, 0, whkF, 0, nullptr, 0, 0, nullptr, 0,
         (int)BS, Dz, Dz, Dz, Dz, Dz, 1);
    attend_kernel<<<dim3(Bz), dim3(256), 0, stream>>>(
        g1, whkF, emb2vf, wa_w, out_w, out_b, out_f + 2 * Bz);
}
